// MultiHeadAttention_3152505995739
// MI455X (gfx1250) — matrix operand order repaired and checked by reading, not compiled, not run on hardware
//
#include <hip/hip_runtime.h>

// MHA forward for B=2, S=2048, d_model=1024, H=16, d_k=64 on gfx1250.
// All GEMMs use v_wmma_f32_16x16x32_bf16 (fp32 accumulate); attention is
// flash-style (online softmax, no materialized SxS scores).

#define HEADS   16
#define DMODEL  1024
#define DK      64
#define BATCH   2
#define SLEN    2048
#define NEGV    (-1000000000.0f)

typedef __attribute__((ext_vector_type(16))) __bf16 v16bf;
typedef __attribute__((ext_vector_type(8)))  float  v8f;

union FragU { v16bf bf; unsigned int u[8]; };

__device__ __forceinline__ unsigned short f2bf(float x) {
  unsigned int u = __float_as_uint(x);
  u += 0x7FFFu + ((u >> 16) & 1u);   // round-to-nearest-even
  return (unsigned short)(u >> 16);
}

__device__ __forceinline__ unsigned int pack2bf(float lo, float hi) {
  return (unsigned int)f2bf(lo) | ((unsigned int)f2bf(hi) << 16);
}

// ---- CDNA5 WMMA fragment gathers (16-bit, 16x16x32) ---------------------
// A (16x32, row-major, K contiguous): lane m = lane&15.
// VGPR j<4 : K = half*8 + j*2 {+0,1};  j>=4 : K = 16 + half*8 + (j-4)*2.
// => dword index within row = kDword0 + half*4 + (j&3) + ((j>=4)?8:0)
__device__ __forceinline__ v16bf load_a_frag(const unsigned int* base, int row,
                                             int rowDwords, int kDword0, int lane) {
  FragU f;
  const unsigned int* rp = base + (size_t)row * rowDwords + kDword0 + ((lane >> 4) << 2);
#pragma unroll
  for (int j = 0; j < 8; ++j) f.u[j] = rp[(j & 3) + ((j >> 2) << 3)];
  return f.bf;
}

// B (32x16): lane n = lane&15; same K order as A.
// VGPR j<4 : K = half*8 + j*2 {+0,1};  j>=4 : K = 16 + half*8 + (j-4)*2.
// `base` is row-major [n][k] (i.e. the GEMM's B transposed), K contiguous.
__device__ __forceinline__ v16bf load_b_frag(const unsigned int* base, int row,
                                             int rowDwords, int kDword0, int lane) {
  FragU f;
  const unsigned int* rp = base + (size_t)row * rowDwords + kDword0 + ((lane >> 4) << 2);
#pragma unroll
  for (int j = 0; j < 8; ++j) f.u[j] = rp[(j & 3) + ((j >> 2) << 3)];
  return f.bf;
}

__device__ __forceinline__ v8f wmma_bf16(v16bf a, v16bf b, v8f c) {
  return __builtin_amdgcn_wmma_f32_16x16x32_bf16(false, a, false, b, (short)0, c,
                                                 false, false);
}

// ---- fp32 -> bf16 conversion (4 elements/thread) ------------------------
__global__ __launch_bounds__(256) void cvt_f32_to_bf16(const float* __restrict__ in,
                                                       unsigned short* __restrict__ out,
                                                       int n4) {
  int i = blockIdx.x * 256 + threadIdx.x;
  if (i < n4) {
    float4 f = ((const float4*)in)[i];
    uint2 p;
    p.x = pack2bf(f.x, f.y);
    p.y = pack2bf(f.z, f.w);
    ((uint2*)out)[i] = p;
  }
}

// ---- GEMM: Y = X @ W^T  (X:[4096,1024] bf16, W:[1024,1024] bf16) -------
// MODE 0: write bf16 to [B,H,S,DK]     (Q / K head layout)
// MODE 1: write bf16 to [B,H,DK,S]     (V transposed for P@V; 16B packed stores)
// MODE 2: write fp32 to [B*S, DMODEL]  (final output projection)
template <int MODE>
__global__ __launch_bounds__(128) void gemm_bf16(const unsigned int* __restrict__ A,
                                                 const unsigned int* __restrict__ W,
                                                 void* __restrict__ out) {
  const int lane = threadIdx.x & 31;
  const int wave = threadIdx.x >> 5;
  const int nl = lane & 15, hl = lane >> 4;
  const int m0 = blockIdx.x * 16;
  const int n0 = (blockIdx.y * 4 + wave) * 64;

  v8f acc[4] = {};
#pragma unroll 1
  for (int kd = 0; kd < DMODEL / 32; ++kd) {
    v16bf a = load_a_frag(A, m0 + nl, DMODEL / 2, kd * 16, lane);
#pragma unroll
    for (int t = 0; t < 4; ++t) {
      v16bf b = load_b_frag(W, n0 + t * 16 + nl, DMODEL / 2, kd * 16, lane);
      acc[t] = wmma_bf16(a, b, acc[t]);
    }
  }

  // C/D layout: VGPR r holds row m = r + hl*8, column n = nl (within tile).
  const int bb = m0 >> 11;               // batch (m0 is 16-aligned, S=2048)
  const int s0 = (m0 & (SLEN - 1)) + hl * 8;

  if (MODE == 1) {
    // lane's 8 rows are consecutive s at fixed d -> one 16B store per tile
#pragma unroll
    for (int t = 0; t < 4; ++t) {
      int gn = n0 + t * 16 + nl;
      int h = gn >> 6, d = gn & (DK - 1);
      uint4 p;
      p.x = pack2bf(acc[t][0], acc[t][1]);
      p.y = pack2bf(acc[t][2], acc[t][3]);
      p.z = pack2bf(acc[t][4], acc[t][5]);
      p.w = pack2bf(acc[t][6], acc[t][7]);
      *(uint4*)((unsigned short*)out +
                (((size_t)(bb * HEADS + h)) * DK + d) * SLEN + s0) = p;
    }
  } else {
#pragma unroll
    for (int t = 0; t < 4; ++t) {
      int gn = n0 + t * 16 + nl;
#pragma unroll
      for (int r = 0; r < 8; ++r) {
        float v = acc[t][r];
        if (MODE == 2) {
          ((float*)out)[((size_t)(m0 + r + hl * 8)) * DMODEL + gn] = v;
        } else {
          int h = gn >> 6, d = gn & (DK - 1);
          ((unsigned short*)out)[(((size_t)(bb * HEADS + h)) * SLEN + s0 + r) * DK + d] =
              f2bf(v);
        }
      }
    }
  }
}

// ---- Flash attention: one wave per (b, h, 16-row q tile) ----------------
__global__ __launch_bounds__(128) void attn_flash(const unsigned int* __restrict__ Qh,
                                                  const unsigned int* __restrict__ Kh,
                                                  const unsigned int* __restrict__ Vt,
                                                  const unsigned char* __restrict__ mask,
                                                  unsigned short* __restrict__ Oh) {
  __shared__ unsigned int Plds[4][16 * 16];   // per-wave 16x32 bf16 P staging

  const int lane = threadIdx.x & 31;
  const int wave = threadIdx.x >> 5;
  const int nl = lane & 15, hl = lane >> 4;

  const int wid = blockIdx.x * 4 + wave;      // 4096 waves total
  const int b  = wid >> 11;                   // / (HEADS * SLEN/16)
  const int h  = (wid >> 7) & (HEADS - 1);
  const int q0 = (wid & 127) << 4;

  const unsigned int* Qb = Qh + (size_t)(b * HEADS + h) * SLEN * (DK / 2);
  const unsigned int* Kb = Kh + (size_t)(b * HEADS + h) * SLEN * (DK / 2);
  const unsigned int* Vb = Vt + (size_t)(b * HEADS + h) * DK * (SLEN / 2);
  const unsigned char* mrow = mask + (size_t)b * SLEN * SLEN;

  // Persistent Q fragments: 16x64 = two 16x32 A fragments.
  v16bf qf0 = load_a_frag(Qb, q0 + nl, DK / 2, 0, lane);
  v16bf qf1 = load_a_frag(Qb, q0 + nl, DK / 2, 16, lane);

  float mrun[8], lrun[8];
#pragma unroll
  for (int r = 0; r < 8; ++r) { mrun[r] = -3.0e38f; lrun[r] = 0.0f; }
  v8f acc[4] = {};

#pragma unroll 1
  for (int k0 = 0; k0 < SLEN; k0 += 32) {
    // --- scores: S = Q(16x64) @ K_block^T -> two 16x16 tiles -------------
    v8f sc[2] = {};
#pragma unroll
    for (int t = 0; t < 2; ++t) {
      int krow = k0 + t * 16 + nl;
      sc[t] = wmma_bf16(qf0, load_b_frag(Kb, krow, DK / 2, 0, lane), sc[t]);
      sc[t] = wmma_bf16(qf1, load_b_frag(Kb, krow, DK / 2, 16, lane), sc[t]);
    }

    // --- scale, mask, online softmax ------------------------------------
    float s0[8], s1[8], bm[8];
#pragma unroll
    for (int r = 0; r < 8; ++r) {
      int row = q0 + r + hl * 8;
      float a0 = sc[0][r] * 0.125f;                   // 1/sqrt(64)
      float a1 = sc[1][r] * 0.125f;
      if (mrow[(size_t)row * SLEN + k0 + nl])      a0 = NEGV;
      if (mrow[(size_t)row * SLEN + k0 + 16 + nl]) a1 = NEGV;
      s0[r] = a0; s1[r] = a1;
      float ml = fmaxf(a0, a1);                       // row max over 16 lanes
#pragma unroll
      for (int off = 1; off < 16; off <<= 1) ml = fmaxf(ml, __shfl_xor(ml, off));
      bm[r] = ml;
    }

    unsigned short* Pw = (unsigned short*)&Plds[wave][0];
#pragma unroll
    for (int r = 0; r < 8; ++r) {
      float mn = fmaxf(mrun[r], bm[r]);
      float alpha = __expf(mrun[r] - mn);
      mrun[r] = mn;
      float p0 = __expf(s0[r] - mn);
      float p1 = __expf(s1[r] - mn);
      float ps = p0 + p1;
#pragma unroll
      for (int off = 1; off < 16; off <<= 1) ps += __shfl_xor(ps, off);
      lrun[r] = lrun[r] * alpha + ps;
#pragma unroll
      for (int t = 0; t < 4; ++t) acc[t][r] = acc[t][r] * alpha;
      // stage P (C-layout -> row-major 16x32 bf16 in LDS)
      int m = r + hl * 8;
      Pw[m * 32 + 0 * 16 + nl] = f2bf(p0);
      Pw[m * 32 + 1 * 16 + nl] = f2bf(p1);
    }
    asm volatile("" ::: "memory");  // keep DS stores before DS loads

    // --- O += P(16x32) @ V_block(32x64) ---------------------------------
    v16bf pf = load_a_frag(&Plds[wave][0], nl, 16, 0, lane);
#pragma unroll
    for (int t = 0; t < 4; ++t) {
      v16bf vf = load_b_frag(Vb, t * 16 + nl, SLEN / 2, k0 / 2, lane);
      acc[t] = wmma_bf16(pf, vf, acc[t]);
    }
  }

  // --- normalize and store bf16 to Oh[B*S, DMODEL] -----------------------
#pragma unroll
  for (int r = 0; r < 8; ++r) {
    float inv = 1.0f / lrun[r];
    int row = q0 + r + hl * 8;
    size_t ob = ((size_t)b * SLEN + row) * DMODEL + h * DK;
#pragma unroll
    for (int t = 0; t < 4; ++t)
      Oh[ob + t * 16 + nl] = f2bf(acc[t][r] * inv);
  }
}

// ------------------------------------------------------------------------
extern "C" void kernel_launch(void* const* d_in, const int* in_sizes, int n_in,
                              void* d_out, int out_size, void* d_ws, size_t ws_size,
                              hipStream_t stream) {
  const float* q  = (const float*)d_in[0];
  const float* k  = (const float*)d_in[1];
  const float* v  = (const float*)d_in[2];
  const unsigned char* mask = (const unsigned char*)d_in[3];  // jnp bool = 1 byte
  const float* Wq = (const float*)d_in[4];
  const float* Wk = (const float*)d_in[5];
  const float* Wv = (const float*)d_in[6];
  const float* Wo = (const float*)d_in[7];

  const int nX = BATCH * SLEN * DMODEL;   // 4,194,304
  const int nW = DMODEL * DMODEL;         // 1,048,576

  // Workspace layout (bf16 halves), total 64 MB.
  unsigned short* Xq  = (unsigned short*)d_ws;
  unsigned short* Xk  = Xq  + nX;
  unsigned short* Xv  = Xk  + nX;
  unsigned short* Wqb = Xv  + nX;
  unsigned short* Wkb = Wqb + nW;
  unsigned short* Wvb = Wkb + nW;
  unsigned short* Wob = Wvb + nW;
  unsigned short* Qh  = Wob + nW;         // [B,H,S,DK]
  unsigned short* Kh  = Qh  + nX;         // [B,H,S,DK]
  unsigned short* Vt  = Kh  + nX;         // [B,H,DK,S]
  unsigned short* Oh  = Vt  + nX;         // [B*S, DMODEL]

  // 1) fp32 -> bf16 conversions (4 elems/thread)
  cvt_f32_to_bf16<<<(nX / 4 + 255) / 256, 256, 0, stream>>>(q,  Xq,  nX / 4);
  cvt_f32_to_bf16<<<(nX / 4 + 255) / 256, 256, 0, stream>>>(k,  Xk,  nX / 4);
  cvt_f32_to_bf16<<<(nX / 4 + 255) / 256, 256, 0, stream>>>(v,  Xv,  nX / 4);
  cvt_f32_to_bf16<<<(nW / 4 + 255) / 256, 256, 0, stream>>>(Wq, Wqb, nW / 4);
  cvt_f32_to_bf16<<<(nW / 4 + 255) / 256, 256, 0, stream>>>(Wk, Wkb, nW / 4);
  cvt_f32_to_bf16<<<(nW / 4 + 255) / 256, 256, 0, stream>>>(Wv, Wvb, nW / 4);
  cvt_f32_to_bf16<<<(nW / 4 + 255) / 256, 256, 0, stream>>>(Wo, Wob, nW / 4);

  // 2) projections (WMMA GEMMs)
  dim3 gg(BATCH * SLEN / 16, DMODEL / 256);   // 256 x 4 blocks, 128 thr (4 waves)
  gemm_bf16<0><<<gg, 128, 0, stream>>>((const unsigned int*)Xq, (const unsigned int*)Wqb, Qh);
  gemm_bf16<0><<<gg, 128, 0, stream>>>((const unsigned int*)Xk, (const unsigned int*)Wkb, Kh);
  gemm_bf16<1><<<gg, 128, 0, stream>>>((const unsigned int*)Xv, (const unsigned int*)Wvb, Vt);

  // 3) flash attention: 4096 waves = 1024 blocks x 4 waves
  attn_flash<<<BATCH * HEADS * (SLEN / 16) / 4, 128, 0, stream>>>(
      (const unsigned int*)Qh, (const unsigned int*)Kh, (const unsigned int*)Vt,
      mask, Oh);

  // 4) output projection -> fp32 d_out
  gemm_bf16<2><<<gg, 128, 0, stream>>>((const unsigned int*)Oh, (const unsigned int*)Wob,
                                       d_out);
}
